// GCNN_81295140979190
// MI455X (gfx1250) — matrix operand order repaired and checked by reading, not compiled, not run on hardware
//
#include <hip/hip_runtime.h>
#include <hip/hip_bf16.h>
#include <math.h>

// CDNA5 WMMA fragment types (wave32)
typedef __attribute__((ext_vector_type(16))) _Float16 v16h;
typedef __attribute__((ext_vector_type(8)))  float    v8f;

#define EPSV 1e-9f

// ---------------------------------------------------------------------------
// Fast transcendental helpers -> single v_rcp_f32 / v_exp_f32 / v_log_f32
__device__ __forceinline__ float fast_rcp(float x) { return __builtin_amdgcn_rcpf(x); }
__device__ __forceinline__ float fast_sigmoid(float x) {
    return fast_rcp(1.f + __expf(-x));
}
__device__ __forceinline__ float fast_tanh(float x) {
    float xc = fminf(fmaxf(x, -9.f), 9.f);
    float e = __expf(2.f * xc);
    return (e - 1.f) * fast_rcp(e + 1.f);
}
__device__ __forceinline__ float fast_softplus(float x) {
    return (x > 20.f) ? x : __logf(1.f + __expf(x));
}

// A-fragment K-index map for 16x32 f16 A matrix (contiguous, matches B):
// lanes 0-15 row M=lane: K0..15
// lanes16-31 row M=lane-16: K16..31
__device__ __forceinline__ int jmapA(int e, int hf) {
    return e + hf * 16;
}

// ---------------------------------------------------------------------------
// Pairwise: D2[b,i,j] = |d2|+eps (== D^2 in the Gaussian), Cos[b,i,j]
__global__ void pairwise_kernel(const float* __restrict__ C,
                                float* __restrict__ D2, float* __restrict__ Cos,
                                int Nn) {
    int b = blockIdx.x / Nn;
    int i = blockIdx.x % Nn;
    const float* Cb = C + (size_t)b * Nn * 3;
    float cx = Cb[i * 3 + 0], cy = Cb[i * 3 + 1], cz = Cb[i * 3 + 2];
    float l2i = cx * cx + cy * cy + cz * cz;
    float ni = sqrtf(l2i + EPSV);
    size_t rowoff = ((size_t)b * Nn + i) * Nn;
    for (int j = threadIdx.x; j < Nn; j += blockDim.x) {
        float dx = Cb[j * 3 + 0], dy = Cb[j * 3 + 1], dz = Cb[j * 3 + 2];
        float l2j = dx * dx + dy * dy + dz * dz;
        float dot = cx * dx + cy * dy + cz * dz;
        float d2 = l2i + l2j - 2.f * dot;
        D2[rowoff + j] = fabsf(d2) + EPSV;
        Cos[rowoff + j] = dot * fast_rcp(ni * sqrtf(l2j + EPSV));
    }
}

// ---------------------------------------------------------------------------
// u = softplus(V@gkw+gkb), w = sigmoid(V@acw+acb)
template <int Ff>
__global__ void gkac_kernel(const float* __restrict__ V,
                            const float* __restrict__ gkw, const float* __restrict__ gkb,
                            const float* __restrict__ acw, const float* __restrict__ acb,
                            float* __restrict__ U, float* __restrict__ W, int total) {
    int idx = blockIdx.x * blockDim.x + threadIdx.x;
    if (idx >= total) return;
    float au[8], aw[8];
    #pragma unroll
    for (int k = 0; k < 8; ++k) { au[k] = gkb[k]; aw[k] = acb[k]; }
    const float* vrow = V + (size_t)idx * Ff;
    for (int f = 0; f < Ff; ++f) {
        float v = vrow[f];
        #pragma unroll
        for (int k = 0; k < 8; ++k) {
            au[k] = fmaf(v, gkw[f * 8 + k], au[k]);
            aw[k] = fmaf(v, acw[f * 8 + k], aw[k]);
        }
    }
    #pragma unroll
    for (int k = 0; k < 8; ++k) {
        U[(size_t)idx * 8 + k] = fast_softplus(au[k]);
        W[(size_t)idx * 8 + k] = fast_sigmoid(aw[k]);
    }
}

// ---------------------------------------------------------------------------
// Fused message passing: M[b, i, k*Ff+f] = sum_j A[b,k,i,j] * V[b,j,f]
// A on the fly: exp(-D2[i,j]*u[i,k]) * (w[i,k]*cos[i,j] + 1-w[i,k])
// Block = 16-row i tile; 8 waves, wave <k> owns graph-kernel k, all Ff cols.
// LDS layouts chosen so each lane's fragment elements are contiguous and rows
// are 16B-aligned with conflict-free bank striding (36 f32 / 56 f16 strides).
template <int Nn, int Ff>
__global__ void __launch_bounds__(256)
msgpass_kernel(const float* __restrict__ D2, const float* __restrict__ Cos,
               const float* __restrict__ U, const float* __restrict__ W,
               const float* __restrict__ V, float* __restrict__ M) {
    constexpr int NT = Nn / 16;
    constexpr int NF = Ff / 16;
    constexpr int LDM = 8 * Ff;
    int b = blockIdx.x / NT;
    int i0 = (blockIdx.x % NT) * 16;
    int tid = threadIdx.x;
    int k = tid >> 5;          // wave id == graph-kernel id
    int lane = tid & 31;
    int hf = lane >> 4;
    int mrow = lane & 15;

    __shared__ __align__(16) float d2t[16][36];       // 144B rows: aligned, conflict-free
    __shared__ __align__(16) float cst[16][36];
    __shared__ __align__(16) _Float16 vtT[Ff][56];    // transposed: [f][j], 112B rows

    float u_mk = U[(((size_t)b * Nn) + i0 + mrow) * 8 + k];
    float w_mk = W[(((size_t)b * Nn) + i0 + mrow) * 8 + k];
    float omw = 1.f - w_mk;

    v8f acc[NF];
    #pragma unroll
    for (int t = 0; t < NF; ++t)
        #pragma unroll
        for (int r = 0; r < 8; ++r) acc[t][r] = 0.f;

    for (int j0 = 0; j0 < Nn; j0 += 32) {
        __syncthreads();
        #pragma unroll
        for (int l = 0; l < 2; ++l) {               // d2/cos tiles: 512 f32 each
            int idx = tid + l * 256;
            int r = idx >> 5, c = idx & 31;
            size_t g = (((size_t)b * Nn) + i0 + r) * Nn + j0 + c;
            d2t[r][c] = D2[g];
            cst[r][c] = Cos[g];
        }
        constexpr int VE = 32 * Ff / 256;
        #pragma unroll
        for (int l = 0; l < VE; ++l) {              // V tile, f32 -> f16, transposed
            int idx = tid + l * 256;
            int jr = idx / Ff, f = idx % Ff;
            vtT[f][jr] = (_Float16)V[(((size_t)b * Nn) + j0 + jr) * Ff + f];
        }
        __syncthreads();

        v16h afrag;
        #pragma unroll
        for (int e = 0; e < 16; ++e) {
            int jl = jmapA(e, hf);                  // one contiguous 16-float run
            float a = __expf(-d2t[mrow][jl] * u_mk) * fmaf(w_mk, cst[mrow][jl], omw);
            afrag[e] = (_Float16)a;
        }
        #pragma unroll
        for (int t = 0; t < NF; ++t) {
            const _Float16* brow = &vtT[t * 16 + mrow][hf * 16];
            v16h bfrag;
            #pragma unroll
            for (int e = 0; e < 16; ++e) bfrag[e] = brow[e];  // contiguous 32B
            acc[t] = __builtin_amdgcn_wmma_f32_16x16x32_f16(
                false, afrag, false, bfrag, (short)0, acc[t], false, false);
        }
    }

    #pragma unroll
    for (int t = 0; t < NF; ++t)
        #pragma unroll
        for (int r = 0; r < 8; ++r) {
            int row = i0 + r + 8 * hf;
            M[(((size_t)b * Nn) + row) * LDM + k * Ff + t * 16 + mrow] = acc[t][r];
        }
}

// ---------------------------------------------------------------------------
// GraphConv dense + tanh + BN(eval) + seq-pool(2)
template <int Nn, int Fin, int FinTot, int Fout, int TPW>
__global__ void __launch_bounds__(256)
densepool_kernel(const float* __restrict__ V, const float* __restrict__ M,
                 const float* __restrict__ Wgt, const float* __restrict__ bias,
                 const float* __restrict__ bng, const float* __restrict__ bnb,
                 float* __restrict__ Out) {
    constexpr int NT = Nn / 16;
    constexpr int LDM = FinTot - Fin;   // 8*Fin
    int b = blockIdx.x / NT;
    int i0 = (blockIdx.x % NT) * 16;
    int tid = threadIdx.x;
    int wave = tid >> 5;
    int lane = tid & 31;
    int hf = lane >> 4;
    int mrow = lane & 15;

    __shared__ __align__(16) _Float16 xt[16][32];     // 64B rows, aligned
    __shared__ __align__(16) _Float16 wtT[Fout][56];  // transposed: [col][k]

    v8f acc[TPW];
    #pragma unroll
    for (int t = 0; t < TPW; ++t)
        #pragma unroll
        for (int r = 0; r < 8; ++r) acc[t][r] = 0.f;

    for (int c0 = 0; c0 < FinTot; c0 += 32) {
        __syncthreads();
        #pragma unroll
        for (int l = 0; l < 2; ++l) {               // X tile (concat on the fly)
            int idx = tid + l * 256;
            int r = idx >> 5, c = idx & 31;
            int col = c0 + c;
            float x = (col < Fin)
                ? V[(((size_t)b * Nn) + i0 + r) * Fin + col]
                : M[(((size_t)b * Nn) + i0 + r) * LDM + (col - Fin)];
            xt[r][c] = (_Float16)x;
        }
        constexpr int WE = 32 * Fout / 256;
        #pragma unroll
        for (int l = 0; l < WE; ++l) {              // weight tile, transposed
            int idx = tid + l * 256;
            int r = idx / Fout, c = idx % Fout;
            wtT[c][r] = (_Float16)Wgt[(size_t)(c0 + r) * Fout + c];
        }
        __syncthreads();

        v16h afrag;
        #pragma unroll
        for (int e = 0; e < 16; ++e) afrag[e] = xt[mrow][jmapA(e, hf)];
        #pragma unroll
        for (int t = 0; t < TPW; ++t) {
            const _Float16* brow = &wtT[(wave * TPW + t) * 16 + mrow][hf * 16];
            v16h bfrag;
            #pragma unroll
            for (int e = 0; e < 16; ++e) bfrag[e] = brow[e];
            acc[t] = __builtin_amdgcn_wmma_f32_16x16x32_f16(
                false, afrag, false, bfrag, (short)0, acc[t], false, false);
        }
    }

    #pragma unroll
    for (int t = 0; t < TPW; ++t) {
        int col = (wave * TPW + t) * 16 + mrow;
        float g = bng[col], bb = bnb[col], bs = bias[col];
        #pragma unroll
        for (int r = 0; r < 8; r += 2) {            // pooled pair lives in one lane
            float h0 = fmaf(g, fast_tanh(acc[t][r] + bs), bb);
            float h1 = fmaf(g, fast_tanh(acc[t][r + 1] + bs), bb);
            int prow = (i0 >> 1) + (r >> 1) + 4 * hf;
            Out[(((size_t)b * (Nn / 2)) + prow) * Fout + col] = 0.5f * (h0 + h1);
        }
    }
}

// ---------------------------------------------------------------------------
__global__ void poolc_kernel(const float* __restrict__ C, float* __restrict__ C1,
                             int total, int N2) {
    int idx = blockIdx.x * blockDim.x + threadIdx.x;
    if (idx >= total) return;
    int c = idx % 3;
    int n = (idx / 3) % N2;
    int b = idx / (3 * N2);
    const float* src = C + (((size_t)b * (2 * N2)) + 2 * n) * 3 + c;
    C1[idx] = 0.5f * (src[0] + src[3]);
}

// ---------------------------------------------------------------------------
// FC head: out[16,512] = sigmoid(Flat[16,32768] @ fc_w + fc_b)
// 16 batch rows == one WMMA M tile. Block covers 128 output cols (8 waves);
// weight chunk staged in LDS with coalesced global reads.
__global__ void __launch_bounds__(256)
fc_kernel(const float* __restrict__ X, const float* __restrict__ Wgt,
          const float* __restrict__ bias, float* __restrict__ Out) {
    constexpr int KD = 32768, FOUT = 512;
    int tid = threadIdx.x;
    int wave = tid >> 5, lane = tid & 31, hf = lane >> 4, mrow = lane & 15;
    int cbase = blockIdx.x * 128;                   // 4 blocks x 128 cols

    __shared__ __align__(16) _Float16 xt[16][32];
    __shared__ __align__(16) _Float16 wtT[128][56]; // [colLocal][k]

    v8f acc;
    #pragma unroll
    for (int r = 0; r < 8; ++r) acc[r] = 0.f;

    for (int k0 = 0; k0 < KD; k0 += 32) {
        __syncthreads();
        #pragma unroll
        for (int l = 0; l < 2; ++l) {               // activation tile 16x32
            int idx = tid + l * 256;
            int r = idx >> 5, c = idx & 31;
            xt[r][c] = (_Float16)X[(size_t)r * KD + k0 + c];
        }
        #pragma unroll
        for (int l = 0; l < 16; ++l) {              // weight tile 32x128, transposed
            int idx = tid + l * 256;
            int r = idx >> 7, c = idx & 127;
            wtT[c][r] = (_Float16)Wgt[(size_t)(k0 + r) * FOUT + cbase + c];
        }
        __syncthreads();
        v16h afrag;
        #pragma unroll
        for (int e = 0; e < 16; ++e) afrag[e] = xt[mrow][jmapA(e, hf)];
        const _Float16* brow = &wtT[wave * 16 + mrow][hf * 16];
        v16h bfrag;
        #pragma unroll
        for (int e = 0; e < 16; ++e) bfrag[e] = brow[e];
        acc = __builtin_amdgcn_wmma_f32_16x16x32_f16(
            false, afrag, false, bfrag, (short)0, acc, false, false);
    }

    int col = cbase + wave * 16 + mrow;
    float bs = bias[col];
    #pragma unroll
    for (int r = 0; r < 8; ++r) {
        int row = r + 8 * hf;
        Out[(size_t)row * FOUT + col] = fast_sigmoid(acc[r] + bs);
    }
}

// ---------------------------------------------------------------------------
extern "C" void kernel_launch(void* const* d_in, const int* in_sizes, int n_in,
                              void* d_out, int out_size, void* d_ws, size_t ws_size,
                              hipStream_t stream) {
    const float* V    = (const float*)d_in[0];
    const float* C    = (const float*)d_in[1];
    const float* gkw0 = (const float*)d_in[2];
    const float* gkb0 = (const float*)d_in[3];
    const float* acw0 = (const float*)d_in[4];
    const float* acb0 = (const float*)d_in[5];
    const float* gcw0 = (const float*)d_in[6];
    const float* gcb0 = (const float*)d_in[7];
    const float* bng0 = (const float*)d_in[8];
    const float* bnb0 = (const float*)d_in[9];
    const float* gkw1 = (const float*)d_in[10];
    const float* gkb1 = (const float*)d_in[11];
    const float* acw1 = (const float*)d_in[12];
    const float* acb1 = (const float*)d_in[13];
    const float* gcw1 = (const float*)d_in[14];
    const float* gcb1 = (const float*)d_in[15];
    const float* bng1 = (const float*)d_in[16];
    const float* bnb1 = (const float*)d_in[17];
    const float* fcw  = (const float*)d_in[18];
    const float* fcb  = (const float*)d_in[19];
    float* out = (float*)d_out;

    // workspace layout (floats); layer-1 reuses layer-0 buffers
    float* ws = (float*)d_ws;
    float* d2 = ws;                   // 16*512*512
    float* cs = d2 + 4194304;         // 16*512*512
    float* U  = cs + 4194304;         // 16*512*8
    float* W  = U + 65536;            // 16*512*8
    float* M  = W + 65536;            // 16*512*512 (== 16*256*1024)
    float* H0 = M + 4194304;          // 16*256*128
    float* C1 = H0 + 524288;          // 16*256*3
    float* H1 = C1 + 12288;           // 16*128*256

    // ----- layer 0 (N=512, F=64) -----
    pairwise_kernel<<<16 * 512, 256, 0, stream>>>(C, d2, cs, 512);
    gkac_kernel<64><<<(16 * 512 + 127) / 128, 128, 0, stream>>>(
        V, gkw0, gkb0, acw0, acb0, U, W, 16 * 512);
    msgpass_kernel<512, 64><<<16 * 32, 256, 0, stream>>>(d2, cs, U, W, V, M);
    densepool_kernel<512, 64, 576, 128, 1><<<16 * 32, 256, 0, stream>>>(
        V, M, gcw0, gcb0, bng0, bnb0, H0);
    poolc_kernel<<<(16 * 256 * 3 + 255) / 256, 256, 0, stream>>>(C, C1, 16 * 256 * 3, 256);

    // ----- layer 1 (N=256, F=128) -----
    pairwise_kernel<<<16 * 256, 256, 0, stream>>>(C1, d2, cs, 256);
    gkac_kernel<128><<<(16 * 256 + 127) / 128, 128, 0, stream>>>(
        H0, gkw1, gkb1, acw1, acb1, U, W, 16 * 256);
    msgpass_kernel<256, 128><<<16 * 16, 256, 0, stream>>>(d2, cs, U, W, H0, M);
    densepool_kernel<256, 128, 1152, 256, 2><<<16 * 16, 256, 0, stream>>>(
        H0, M, gcw1, gcb1, bng1, bnb1, H1);

    // ----- FC head -----
    fc_kernel<<<4, 256, 0, stream>>>(H1, fcw, fcb, out);
}
